// CrossAttention_Reverse_4183298146317
// MI455X (gfx1250) — matrix operand order repaired and checked by reading, not compiled, not run on hardware
//
#include <hip/hip_runtime.h>
#include <hip/hip_bf16.h>

typedef __attribute__((ext_vector_type(16))) _Float16 v16h;
typedef __attribute__((ext_vector_type(8)))  _Float16 v8h;
typedef __attribute__((ext_vector_type(8)))  float    v8f;

// ---- problem constants (fixed by the reference) ----
#define BATCH   4
#define FEA     3
#define CCH     256     // C
#define DOUT    256
#define NHEADS  8
#define HDIM    32      // DOUT / HEADS
#define HH      32
#define WW      32
#define RES     1024    // HH*WW
#define LQ      3072    // FEA*RES
#define MQ      (BATCH*LQ)     // 12288 q rows
#define MKV     (BATCH*RES)    // 4096 k/v rows
#define EPS     1e-5f

// ---------------- WMMA helpers (A/B use contiguous k order; C/D layout per CDNA5 ISA 7.12.2) ----------------
__device__ __forceinline__ v8f wmma_f16(v16h a, v16h b, v8f c) {
  return __builtin_amdgcn_wmma_f32_16x16x32_f16(false, a, false, b, (short)0, c,
                                                false, false);
}

// A fragment: 16x32 f16. base points at row 0 of the 16-row strip, ld in halfs.
// lanes 0-15: row=lane,   halfs = K{k0+0..15}
// lanes16-31: row=lane-16,halfs = K{k0+16..31}
__device__ __forceinline__ v16h load_a(const _Float16* base, int ld, int k0, int lane) {
  int l  = lane & 15;
  int hi = (lane >> 4) & 1;
  const _Float16* p = base + (size_t)l * ld + k0 + hi * 16;
  v8h lo8 = *(const v8h*)(p);
  v8h hi8 = *(const v8h*)(p + 8);
  v16h r;
#pragma unroll
  for (int i = 0; i < 8; ++i) { r[i] = lo8[i]; r[i + 8] = hi8[i]; }
  return r;
}

// B fragment: 32x16 f16 (KxN), supplied as Bt[N][K] row-major.
// lanes 0-15: col n=lane, K = k..k+15 ; lanes 16-31: col n=lane-16, K = k+16..k+31
__device__ __forceinline__ v16h load_b(const _Float16* bt_row0, int ld, int lane) {
  int n  = lane & 15;
  int ko = ((lane >> 4) & 1) * 16;
  return *(const v16h*)(bt_row0 + (size_t)n * ld + ko);
}

// C/D layout: VGPR r, lanes 0-15 -> (M=r, N=lane); lanes 16-31 -> (M=8+r, N=lane-16)

__device__ __forceinline__ float redmax16(float v) {
#pragma unroll
  for (int m = 1; m < 16; m <<= 1) v = fmaxf(v, __shfl_xor(v, m, 32));
  return v;
}
__device__ __forceinline__ float redsum16(float v) {
#pragma unroll
  for (int m = 1; m < 16; m <<= 1) v += __shfl_xor(v, m, 32);
  return v;
}

// ---------------- weight fp32 -> f16 ----------------
__global__ void cvt_f16_kernel(const float* __restrict__ a, _Float16* __restrict__ o, int n) {
  int i = blockIdx.x * 256 + threadIdx.x;
  if (i < n) o[i] = (_Float16)a[i];
}

// ---------------- depthwise conv 3x3 + BN on query tokens ----------------
// query: [B, FEA*RES, C]; out qtok f16 [B*LQ, C]
__global__ void conv_q_kernel(const float* __restrict__ q, const float* __restrict__ w,
                              const float* __restrict__ gam, const float* __restrict__ bet,
                              const float* __restrict__ mu, const float* __restrict__ var,
                              _Float16* __restrict__ qtok) {
  int idx  = blockIdx.x * 256 + threadIdx.x;          // [0, B*FEA*RES*C)
  int c    = idx & 255;
  int rest = idx >> 8;                                 // global token index b*LQ + fea*RES + s
  int s    = rest & 1023;
  int fb   = rest >> 10;                               // b*FEA + fea
  int fea  = fb % FEA;
  int b    = fb / FEA;
  int y = s >> 5, x = s & 31;
  const float* wp = w + (size_t)(fea * CCH + c) * 9;
  const float* qb = q + ((size_t)(b * LQ + fea * RES)) * CCH + c;
  float sum = 0.f;
#pragma unroll
  for (int dy = -1; dy <= 1; ++dy) {
    int yy = y + dy; if (yy < 0 || yy > 31) continue;
#pragma unroll
    for (int dx = -1; dx <= 1; ++dx) {
      int xx = x + dx; if (xx < 0 || xx > 31) continue;
      sum += wp[(dy + 1) * 3 + (dx + 1)] * qb[(size_t)(yy * 32 + xx) * CCH];
    }
  }
  int pc = fea * CCH + c;
  float inv = gam[pc] * rsqrtf(var[pc] + EPS);
  float o = sum * inv + (bet[pc] - mu[pc] * inv);
  qtok[(size_t)rest * CCH + c] = (_Float16)o;
}

// ---------------- depthwise conv 3x3 + BN on x -> k_tok and v_tok ----------------
// x: [B, C, H, W] ; outputs token-major f16 [B*RES, C]
__global__ void conv_kv_kernel(const float* __restrict__ x,
                               const float* __restrict__ wk, const float* __restrict__ gk,
                               const float* __restrict__ bk, const float* __restrict__ mk,
                               const float* __restrict__ vk,
                               const float* __restrict__ wv, const float* __restrict__ gv,
                               const float* __restrict__ bv, const float* __restrict__ mv,
                               const float* __restrict__ vv,
                               _Float16* __restrict__ ktok, _Float16* __restrict__ vtok) {
  int idx  = blockIdx.x * 256 + threadIdx.x;   // [0, B*RES*C)
  int c    = idx & 255;
  int rest = idx >> 8;                         // b*RES + s
  int s    = rest & 1023;
  int b    = rest >> 10;
  int y = s >> 5, x0 = s & 31;
  const float* xb  = x + ((size_t)(b * CCH + c)) * RES;
  const float* wkp = wk + (size_t)c * 9;
  const float* wvp = wv + (size_t)c * 9;
  float sk = 0.f, sv = 0.f;
#pragma unroll
  for (int dy = -1; dy <= 1; ++dy) {
    int yy = y + dy; if (yy < 0 || yy > 31) continue;
#pragma unroll
    for (int dx = -1; dx <= 1; ++dx) {
      int xx = x0 + dx; if (xx < 0 || xx > 31) continue;
      float xv = xb[yy * 32 + xx];
      sk += wkp[(dy + 1) * 3 + (dx + 1)] * xv;
      sv += wvp[(dy + 1) * 3 + (dx + 1)] * xv;
    }
  }
  float ik = gk[c] * rsqrtf(vk[c] + EPS);
  float iv = gv[c] * rsqrtf(vv[c] + EPS);
  ktok[(size_t)rest * CCH + c] = (_Float16)(sk * ik + (bk[c] - mk[c] * ik));
  vtok[(size_t)rest * CCH + c] = (_Float16)(sv * iv + (bv[c] - mv[c] * iv));
}

// ---------------- generic WMMA GEMM: out[M,N] = A[M,K] * Bt[N,K]^T ----------------
// mode 0: f16 row-major out; mode 1: f16, per-batch transposed out[(b*N+n)*rpb + t]
// mode 2: f32 row-major out + bias[n]
__global__ void gemm_kernel(const _Float16* __restrict__ A, const _Float16* __restrict__ Bt,
                            void* __restrict__ out, const float* __restrict__ bias,
                            int M, int N, int K, int mode, int rpb) {
  const int lane = threadIdx.x & 31;
  const int wave = threadIdx.x >> 5;
  const int m0 = blockIdx.x * 64 + wave * 16;
  const int n0 = blockIdx.y * 64;
  if (m0 >= M) return;
  v8f c0 = {}, c1 = {}, c2 = {}, c3 = {};
  const _Float16* arow = A + (size_t)m0 * K;
  for (int k0 = 0; k0 < K; k0 += 32) {
    v16h a  = load_a(arow, K, k0, lane);
    v16h b0 = load_b(Bt + (size_t)(n0 +  0) * K + k0, K, lane);
    v16h b1 = load_b(Bt + (size_t)(n0 + 16) * K + k0, K, lane);
    v16h b2 = load_b(Bt + (size_t)(n0 + 32) * K + k0, K, lane);
    v16h b3 = load_b(Bt + (size_t)(n0 + 48) * K + k0, K, lane);
    c0 = wmma_f16(a, b0, c0);
    c1 = wmma_f16(a, b1, c1);
    c2 = wmma_f16(a, b2, c2);
    c3 = wmma_f16(a, b3, c3);
  }
  const int rbase = m0 + ((lane >> 4) & 1) * 8;
  const int col0  = n0 + (lane & 15);
#pragma unroll
  for (int j = 0; j < 4; ++j) {
    v8f cc = (j == 0) ? c0 : (j == 1) ? c1 : (j == 2) ? c2 : c3;
    int col = col0 + j * 16;
#pragma unroll
    for (int r = 0; r < 8; ++r) {
      int row = rbase + r;
      float v = cc[r];
      if (mode == 0) {
        ((_Float16*)out)[(size_t)row * N + col] = (_Float16)v;
      } else if (mode == 1) {
        int bb = row / rpb, t = row % rpb;
        ((_Float16*)out)[((size_t)bb * N + col) * rpb + t] = (_Float16)v;
      } else {
        ((float*)out)[(size_t)row * N + col] = v + bias[col];
      }
    }
  }
}

// ---------------- flash attention: per (b, head), d = 32 ----------------
// Q: [B*LQ, 256] f16 ; Kt: [B*RES, 256] f16 ; Vt: [B][256][RES] f16 (d-major)
// AO: [B*LQ, 256] f16
__global__ void attn_kernel(const _Float16* __restrict__ Q, const _Float16* __restrict__ Kt,
                            const _Float16* __restrict__ Vt, _Float16* __restrict__ AO) {
  __shared__ alignas(32) _Float16 ldsP[4][16 * 32];
  const int lane = threadIdx.x & 31;
  const int wave = threadIdx.x >> 5;
  const int b = blockIdx.z, h = blockIdx.y;
  const int row0 = blockIdx.x * 64 + wave * 16;
  const float scale = 0.0625f;  // DIM_OUT^-0.5 = 1/16

  const _Float16* qbase = Q  + ((size_t)(b * LQ + row0)) * DOUT + h * HDIM;
  const _Float16* kbase = Kt + ((size_t)(b * RES)) * DOUT + h * HDIM;
  const _Float16* vbase = Vt + ((size_t)b * DOUT + h * HDIM) * RES;
  _Float16* myP = ldsP[wave];

  v16h aq = load_a(qbase, DOUT, 0, lane);  // whole head dim in one fragment

  float mrow[8], lrow[8];
#pragma unroll
  for (int r = 0; r < 8; ++r) { mrow[r] = -3.0e38f; lrow[r] = 0.f; }
  v8f acc0 = {}, acc1 = {};
  const v8f zf = {};
  const int lo16 = lane & 15;
  const int rofs = ((lane >> 4) & 1) * 8;

  for (int t0 = 0; t0 < RES; t0 += 32) {
    // S = Q * K^T for 32 keys (two 16-col tiles)
    v16h bk0 = load_b(kbase + (size_t)(t0 +  0) * DOUT, DOUT, lane);
    v16h bk1 = load_b(kbase + (size_t)(t0 + 16) * DOUT, DOUT, lane);
    v8f s0 = wmma_f16(aq, bk0, zf);
    v8f s1 = wmma_f16(aq, bk1, zf);
    // online softmax per row; write P tile to per-wave LDS in row-major
#pragma unroll
    for (int r = 0; r < 8; ++r) {
      float x0 = s0[r] * scale, x1 = s1[r] * scale;
      float cm = redmax16(fmaxf(x0, x1));
      float mn = fmaxf(mrow[r], cm);
      float alpha = __expf(mrow[r] - mn);
      float p0 = __expf(x0 - mn), p1 = __expf(x1 - mn);
      lrow[r] = lrow[r] * alpha + redsum16(p0 + p1);
      mrow[r] = mn;
      acc0[r] *= alpha; acc1[r] *= alpha;
      int row = r + rofs;
      myP[row * 32 + lo16]      = (_Float16)p0;
      myP[row * 32 + 16 + lo16] = (_Float16)p1;
    }
    asm volatile("s_wait_dscnt 0" ::: "memory");
    // O += P * V  (P: 16x32 A-frag from LDS; Vt rows are d, contiguous in t)
    v16h ap  = load_a(myP, 32, 0, lane);
    v16h bv0 = load_b(vbase + t0,              RES, lane);  // d = 0..15
    v16h bv1 = load_b(vbase + 16 * RES + t0,   RES, lane);  // d = 16..31
    acc0 = wmma_f16(ap, bv0, acc0);
    acc1 = wmma_f16(ap, bv1, acc1);
    asm volatile("s_wait_dscnt 0" ::: "memory");  // LDS reads done before next overwrite
  }

  _Float16* obase = AO + ((size_t)(b * LQ + row0)) * DOUT + h * HDIM;
#pragma unroll
  for (int r = 0; r < 8; ++r) {
    int row = r + rofs;
    float inv = 1.0f / lrow[r];
    obase[(size_t)row * DOUT + lo16]      = (_Float16)(acc0[r] * inv);
    obase[(size_t)row * DOUT + 16 + lo16] = (_Float16)(acc1[r] * inv);
  }
}

// ---------------- host launcher ----------------
extern "C" void kernel_launch(void* const* d_in, const int* in_sizes, int n_in,
                              void* d_out, int out_size, void* d_ws, size_t ws_size,
                              hipStream_t stream) {
  (void)in_sizes; (void)n_in; (void)out_size; (void)ws_size;
  const float* x        = (const float*)d_in[0];
  const float* query    = (const float*)d_in[1];
  const float* conv_q_w = (const float*)d_in[2];
  const float* bn_q_g   = (const float*)d_in[3];
  const float* bn_q_b   = (const float*)d_in[4];
  const float* bn_q_m   = (const float*)d_in[5];
  const float* bn_q_v   = (const float*)d_in[6];
  const float* conv_k_w = (const float*)d_in[7];
  const float* bn_k_g   = (const float*)d_in[8];
  const float* bn_k_b   = (const float*)d_in[9];
  const float* bn_k_m   = (const float*)d_in[10];
  const float* bn_k_v   = (const float*)d_in[11];
  const float* conv_v_w = (const float*)d_in[12];
  const float* bn_v_g   = (const float*)d_in[13];
  const float* bn_v_b   = (const float*)d_in[14];
  const float* bn_v_m   = (const float*)d_in[15];
  const float* bn_v_v   = (const float*)d_in[16];
  const float* proj_q_w = (const float*)d_in[17];
  const float* proj_k_w = (const float*)d_in[18];
  const float* proj_v_w = (const float*)d_in[19];
  const float* proj_w   = (const float*)d_in[20];
  const float* proj_b   = (const float*)d_in[21];

  _Float16* ws = (_Float16*)d_ws;
  size_t off = 0;
  _Float16* qtok = ws + off; off += (size_t)MQ  * CCH;   // 3.1M halfs
  _Float16* ktok = ws + off; off += (size_t)MKV * CCH;
  _Float16* vtok = ws + off; off += (size_t)MKV * CCH;
  _Float16* Qp   = ws + off; off += (size_t)MQ  * DOUT;
  _Float16* Kp   = ws + off; off += (size_t)MKV * DOUT;
  _Float16* VTp  = ws + off; off += (size_t)MKV * DOUT;  // stored [B][DOUT][RES]
  _Float16* AOp  = ws + off; off += (size_t)MQ  * DOUT;
  _Float16* WQ   = ws + off; off += (size_t)DOUT * CCH;
  _Float16* WK   = ws + off; off += (size_t)DOUT * CCH;
  _Float16* WV   = ws + off; off += (size_t)DOUT * CCH;
  _Float16* WP   = ws + off; off += (size_t)DOUT * DOUT;

  const int NW = DOUT * CCH;  // 65536
  cvt_f16_kernel<<<NW / 256, 256, 0, stream>>>(proj_q_w, WQ, NW);
  cvt_f16_kernel<<<NW / 256, 256, 0, stream>>>(proj_k_w, WK, NW);
  cvt_f16_kernel<<<NW / 256, 256, 0, stream>>>(proj_v_w, WV, NW);
  cvt_f16_kernel<<<NW / 256, 256, 0, stream>>>(proj_w,   WP, NW);

  conv_q_kernel<<<(MQ * CCH) / 256, 256, 0, stream>>>(
      query, conv_q_w, bn_q_g, bn_q_b, bn_q_m, bn_q_v, qtok);
  conv_kv_kernel<<<(MKV * CCH) / 256, 256, 0, stream>>>(
      x, conv_k_w, bn_k_g, bn_k_b, bn_k_m, bn_k_v,
         conv_v_w, bn_v_g, bn_v_b, bn_v_m, bn_v_v, ktok, vtok);

  // projections
  gemm_kernel<<<dim3(MQ / 64, DOUT / 64), 128, 0, stream>>>(
      qtok, WQ, Qp, nullptr, MQ, DOUT, CCH, 0, 0);
  gemm_kernel<<<dim3(MKV / 64, DOUT / 64), 128, 0, stream>>>(
      ktok, WK, Kp, nullptr, MKV, DOUT, CCH, 0, 0);
  gemm_kernel<<<dim3(MKV / 64, DOUT / 64), 128, 0, stream>>>(
      vtok, WV, VTp, nullptr, MKV, DOUT, CCH, 1, RES);  // transposed per batch

  // attention
  attn_kernel<<<dim3(LQ / 64, NHEADS, BATCH), 128, 0, stream>>>(Qp, Kp, VTp, AOp);

  // final projection + bias -> fp32 output
  gemm_kernel<<<dim3(MQ / 64, DOUT / 64), 128, 0, stream>>>(
      AOp, WP, d_out, proj_b, MQ, DOUT, CCH, 2, 0);
}
